// MultiLabelClassificationGNN_47845935677475
// MI455X (gfx1250) — compile-verified, matrix operand order checked by reading, not run on hardware
//
#include <hip/hip_runtime.h>
#include <stdint.h>

#define NNODES 20000
#define INDIM  256
#define HIDC   128
#define NCLS   64
#define NEDGES 320000

typedef __bf16 bf16_t;
typedef bf16_t v16bf __attribute__((ext_vector_type(16)));
typedef float  v8f   __attribute__((ext_vector_type(8)));

#define BM 128
#define BN 128
#define BK 32
#define LSTRIDE (BK + 8)   // bf16 elements per LDS row (pad vs bank conflicts)

__device__ __forceinline__ bf16_t f2bf(float f) { return (bf16_t)f; }

// ---------------------------------------------------------------------------
// Generic tiled GEMM: C[M,Nn] = act(A[M,K] @ B[K,Nn] + bias)
// f32 inputs staged to LDS as bf16, V_WMMA_F32_16X16X32_BF16, f32 accum/output.
// Block tile 128x128x32; 8 waves in 4(M)x2(N) grid; each wave 2x4 16x16 frags.
// K-tile global loads are register double-buffered (issued before the WMMA
// phase of the previous tile) and branchless (clamped address + cndmask-zero)
// so all 8 global_load_b128 issue back-to-back and overlap the matrix math.
// act: 0 = none, 1 = relu, 2 = sigmoid
// ---------------------------------------------------------------------------
__global__ __launch_bounds__(256) void gemm_bf16_wmma(
    const float* __restrict__ A, const float* __restrict__ B,
    const float* __restrict__ bias, float* __restrict__ C,
    int M, int Nn, int K, int act)
{
  __shared__ bf16_t As[BM][LSTRIDE];   // [row][k], k contiguous
  __shared__ bf16_t Bs[BN][LSTRIDE];   // transposed: [col][k], k contiguous

  const int tid  = threadIdx.x;
  const int lane = tid & 31;
  const int wave = tid >> 5;   // 0..7
  const int wm   = wave & 3;   // 4 waves along M
  const int wn   = wave >> 2;  // 2 waves along N

  const int blockRow = blockIdx.y * BM;
  const int blockCol = blockIdx.x * BN;

  const int hlf = lane >> 4;   // K-half select (ISA 16-bit A/B layout)
  const int mn  = lane & 15;   // M (for A) / N (for B) within fragment

  v8f acc[2][4];
  const v8f vzero = {0.f,0.f,0.f,0.f,0.f,0.f,0.f,0.f};
  for (int i = 0; i < 2; ++i)
    for (int j = 0; j < 4; ++j) acc[i][j] = vzero;

  float4 aR[4], bR[4];

  // Branchless tile loaders: clamp address into range, select zero when OOB.
  auto loadA = [&](int k0) {
    #pragma unroll
    for (int it = 0; it < 4; ++it) {
      int lin = tid + it * 256;       // 0..1023 float4 slots
      int r   = lin >> 3;             // row 0..127
      int g   = lin & 7;              // k-group of 4
      int gr  = blockRow + r;
      int cr  = gr < M ? gr : M - 1;  // address-safe clamp
      float4 v = *(const float4*)(A + (size_t)cr * K + k0 + g * 4);
      if (gr >= M) v = make_float4(0.f, 0.f, 0.f, 0.f);  // v_cndmask, no branch
      aR[it] = v;
    }
  };
  auto loadB = [&](int k0) {
    #pragma unroll
    for (int it = 0; it < 4; ++it) {
      int lin = tid + it * 256;
      int kk  = lin >> 5;             // 0..31
      int cg  = lin & 31;             // col group of 4
      int gc  = blockCol + cg * 4;
      int cc  = gc < Nn ? gc : Nn - 4;
      float4 v = *(const float4*)(B + (size_t)(k0 + kk) * Nn + cc);
      if (gc >= Nn) v = make_float4(0.f, 0.f, 0.f, 0.f);
      bR[it] = v;
    }
  };

  loadA(0);
  loadB(0);

  for (int k0 = 0; k0 < K; k0 += BK) {
    // --- convert current tile regs -> bf16 LDS ---
    #pragma unroll
    for (int it = 0; it < 4; ++it) {
      int lin = tid + it * 256;
      int r   = lin >> 3;
      int g   = lin & 7;
      As[r][g*4+0] = f2bf(aR[it].x); As[r][g*4+1] = f2bf(aR[it].y);
      As[r][g*4+2] = f2bf(aR[it].z); As[r][g*4+3] = f2bf(aR[it].w);
    }
    #pragma unroll
    for (int it = 0; it < 4; ++it) {
      int lin = tid + it * 256;
      int kk  = lin >> 5;
      int cg  = lin & 31;
      Bs[cg*4+0][kk] = f2bf(bR[it].x); Bs[cg*4+1][kk] = f2bf(bR[it].y);
      Bs[cg*4+2][kk] = f2bf(bR[it].z); Bs[cg*4+3][kk] = f2bf(bR[it].w);
    }
    __syncthreads();

    // --- prefetch next K-tile while WMMAs below execute ---
    if (k0 + BK < K) { loadA(k0 + BK); loadB(k0 + BK); }

    // --- build fragments per ISA 7.12.2 (16-bit A 16x32 layout; B mirrored) ---
    v16bf afrag[2], bfrag[4];
    #pragma unroll
    for (int fm = 0; fm < 2; ++fm) {
      int m = wm * 32 + fm * 16 + mn;
      uint32_t* d = (uint32_t*)&afrag[fm];
      #pragma unroll
      for (int v = 0; v < 8; ++v) {
        int kk = ((v & 3) << 1) + (hlf << 3) + ((v >= 4) ? 16 : 0);
        d[v] = *(const uint32_t*)&As[m][kk];   // two consecutive K bf16s
      }
    }
    #pragma unroll
    for (int fn = 0; fn < 4; ++fn) {
      int n = wn * 64 + fn * 16 + mn;
      uint32_t* d = (uint32_t*)&bfrag[fn];
      #pragma unroll
      for (int v = 0; v < 8; ++v) {
        int kk = ((v & 3) << 1) + (hlf << 3) + ((v >= 4) ? 16 : 0);
        d[v] = *(const uint32_t*)&Bs[n][kk];
      }
    }

    #pragma unroll
    for (int fm = 0; fm < 2; ++fm)
      #pragma unroll
      for (int fn = 0; fn < 4; ++fn)
        acc[fm][fn] = __builtin_amdgcn_wmma_f32_16x16x32_bf16(
            false, afrag[fm], false, bfrag[fn],
            (short)0, acc[fm][fn], false, false);
    __syncthreads();
  }

  // --- epilogue: C layout = VGPR i holds M=i+8*half, N=lane&15 ---
  for (int fm = 0; fm < 2; ++fm) {
    for (int fn = 0; fn < 4; ++fn) {
      int col = blockCol + wn * 64 + fn * 16 + mn;
      if (col >= Nn) continue;
      float bv = bias ? bias[col] : 0.0f;
      #pragma unroll
      for (int i = 0; i < 8; ++i) {
        int row = blockRow + wm * 32 + fm * 16 + hlf * 8 + i;
        if (row >= M) continue;
        float x = acc[fm][fn][i] + bv;
        if (act == 1)      x = fmaxf(x, 0.0f);
        else if (act == 2) x = 1.0f / (1.0f + __expf(-x));
        C[(size_t)row * Nn + col] = x;
      }
    }
  }
}

// ---------------------------------------------------------------------------
// GAT helper kernels
// ---------------------------------------------------------------------------
__device__ __forceinline__ float atomicMaxFloat(float* addr, float value) {
  // Hardware int/uint atomics implement float max (bits are order-preserving)
  return (value >= 0.0f)
      ? __int_as_float(atomicMax((int*)addr, __float_as_int(value)))
      : __uint_as_float(atomicMin((unsigned int*)addr, __float_as_uint(value)));
}

template <int HEADS>
__global__ void gat_alpha(const float* __restrict__ H,
                          const float* __restrict__ a_src,
                          const float* __restrict__ a_dst,
                          float* __restrict__ as_, float* __restrict__ ad_,
                          float* __restrict__ mx, float* __restrict__ dn, int N)
{
  int i = blockIdx.x * blockDim.x + threadIdx.x;  // node*HEADS + h
  if (i >= N * HEADS) return;
  int n = i / HEADS, h = i % HEADS;
  const float* hp = H + (size_t)n * (HEADS * HIDC) + h * HIDC;
  const float* sp = a_src + h * HIDC;
  const float* dp = a_dst + h * HIDC;
  float ssum = 0.f, dsum = 0.f;
  for (int c = 0; c < HIDC; c += 4) {
    float4 hv = *(const float4*)(hp + c);
    float4 sv = *(const float4*)(sp + c);
    float4 dv = *(const float4*)(dp + c);
    ssum += hv.x*sv.x + hv.y*sv.y + hv.z*sv.z + hv.w*sv.w;
    dsum += hv.x*dv.x + hv.y*dv.y + hv.z*dv.z + hv.w*dv.w;
  }
  as_[i] = ssum; ad_[i] = dsum;
  mx[i] = -INFINITY; dn[i] = 0.0f;
}

template <int HEADS>
__global__ void gat_edge_max(const int* __restrict__ ei, int E, int N,
                             const float* __restrict__ as_,
                             const float* __restrict__ ad_,
                             float* __restrict__ mx)
{
  int e = blockIdx.x * blockDim.x + threadIdx.x;
  if (e >= E + N) return;
  int s, d;
  if (e < E) { s = ei[e]; d = ei[E + e]; } else { s = d = e - E; }
  #pragma unroll
  for (int h = 0; h < HEADS; ++h) {
    float v = as_[s * HEADS + h] + ad_[d * HEADS + h];
    v = v > 0.f ? v : 0.2f * v;                    // leaky_relu(0.2)
    atomicMaxFloat(&mx[d * HEADS + h], v);
  }
}

__global__ void gat_fix_max(float* __restrict__ mx, int count) {
  int i = blockIdx.x * blockDim.x + threadIdx.x;
  if (i >= count) return;
  float v = mx[i];
  if (!__builtin_isfinite(v)) mx[i] = 0.0f;
}

template <int HEADS>
__global__ void gat_edge_sum(const int* __restrict__ ei, int E, int N,
                             const float* __restrict__ as_,
                             const float* __restrict__ ad_,
                             const float* __restrict__ mx,
                             float* __restrict__ dn)
{
  int e = blockIdx.x * blockDim.x + threadIdx.x;
  if (e >= E + N) return;
  int s, d;
  if (e < E) { s = ei[e]; d = ei[E + e]; } else { s = d = e - E; }
  #pragma unroll
  for (int h = 0; h < HEADS; ++h) {
    float v = as_[s * HEADS + h] + ad_[d * HEADS + h];
    v = v > 0.f ? v : 0.2f * v;
    atomicAdd(&dn[d * HEADS + h], __expf(v - mx[d * HEADS + h]));
  }
}

// One wave per edge; each lane handles 4 channels (float4); head-mean folded in.
template <int HEADS>
__global__ __launch_bounds__(256) void gat_aggregate(
    const float* __restrict__ H, const int* __restrict__ ei, int E, int N,
    const float* __restrict__ as_, const float* __restrict__ ad_,
    const float* __restrict__ mx, const float* __restrict__ dn,
    float* __restrict__ agg)
{
  int edge = blockIdx.x * (blockDim.x >> 5) + (threadIdx.x >> 5);
  int lane = threadIdx.x & 31;
  if (edge >= E + N) return;
  int s, d;
  if (edge < E) { s = ei[edge]; d = ei[E + edge]; } else { s = d = edge - E; }

  float alpha[HEADS];
  #pragma unroll
  for (int h = 0; h < HEADS; ++h) {
    float v = as_[s * HEADS + h] + ad_[d * HEADS + h];
    v = v > 0.f ? v : 0.2f * v;
    float ex = __expf(v - mx[d * HEADS + h]);
    alpha[h] = ex / (dn[d * HEADS + h] + 1e-16f) * (1.0f / HEADS);
  }
  int c = lane * 4;
  float4 acc = make_float4(0.f, 0.f, 0.f, 0.f);
  #pragma unroll
  for (int h = 0; h < HEADS; ++h) {
    float4 v = *(const float4*)(H + (size_t)s * (HEADS * HIDC) + h * HIDC + c);
    acc.x += alpha[h] * v.x; acc.y += alpha[h] * v.y;
    acc.z += alpha[h] * v.z; acc.w += alpha[h] * v.w;
  }
  float* o = agg + (size_t)d * HIDC + c;
  atomicAdd(o + 0, acc.x); atomicAdd(o + 1, acc.y);
  atomicAdd(o + 2, acc.z); atomicAdd(o + 3, acc.w);
}

// y = (do_bn ? relu(g*((x+bias)-mean)*rsqrt(var+eps)+beta) : x+bias); in-place ok
__global__ void bias_bn_relu(const float* __restrict__ x,
                             const float* __restrict__ bias,
                             const float* __restrict__ g,
                             const float* __restrict__ beta,
                             const float* __restrict__ mean,
                             const float* __restrict__ var,
                             float* __restrict__ y, int N, int C, int do_bn)
{
  int i = blockIdx.x * blockDim.x + threadIdx.x;
  if (i >= N * C) return;
  int c = i % C;
  float v = x[i] + bias[c];
  if (do_bn) {
    v = g[c] * (v - mean[c]) * rsqrtf(var[c] + 1e-5f) + beta[c];
    v = fmaxf(v, 0.0f);
  }
  y[i] = v;
}

// ---------------------------------------------------------------------------
extern "C" void kernel_launch(void* const* d_in, const int* in_sizes, int n_in,
                              void* d_out, int out_size, void* d_ws, size_t ws_size,
                              hipStream_t stream) {
  const float* x      = (const float*)d_in[0];
  const int*   ei     = (const int*)d_in[1];
  const float* W1     = (const float*)d_in[2];
  const float* asrc1  = (const float*)d_in[3];
  const float* adst1  = (const float*)d_in[4];
  const float* b1     = (const float*)d_in[5];
  const float* W2     = (const float*)d_in[6];
  const float* asrc2  = (const float*)d_in[7];
  const float* adst2  = (const float*)d_in[8];
  const float* b2     = (const float*)d_in[9];
  const float* W3     = (const float*)d_in[10];
  const float* asrc3  = (const float*)d_in[11];
  const float* adst3  = (const float*)d_in[12];
  const float* b3     = (const float*)d_in[13];
  const float* bn1g   = (const float*)d_in[14];
  const float* bn1b   = (const float*)d_in[15];
  const float* bn1m   = (const float*)d_in[16];
  const float* bn1v   = (const float*)d_in[17];
  const float* bn2g   = (const float*)d_in[18];
  const float* bn2b   = (const float*)d_in[19];
  const float* bn2m   = (const float*)d_in[20];
  const float* bn2v   = (const float*)d_in[21];
  const float* l1w    = (const float*)d_in[22];
  const float* l1b    = (const float*)d_in[23];
  const float* l2w    = (const float*)d_in[24];
  const float* l2b    = (const float*)d_in[25];
  const float* l3w    = (const float*)d_in[26];
  const float* l3b    = (const float*)d_in[27];
  float* out = (float*)d_out;

  const int N = NNODES, E = NEDGES;
  float* ws  = (float*)d_ws;
  float* H   = ws;                         // N*512 multi-head features / scratch
  float* F   = H + (size_t)N * 512;       // N*128 node features (ping)
  float* G   = F + (size_t)N * HIDC;      // N*128 node features (pong)
  float* As_ = G + (size_t)N * HIDC;      // N*4 alpha_src
  float* Ad_ = As_ + (size_t)N * 4;       // N*4 alpha_dst
  float* Mx  = Ad_ + (size_t)N * 4;       // N*4 segment max
  float* Dn  = Mx  + (size_t)N * 4;       // N*4 segment denom

  dim3 blk(256);
  const int totE   = E + N;
  const int edgeBl = (totE + 255) / 256;
  const int aggBl  = (totE + 7) / 8;      // 8 edges per 256-thread block

  #define GEMM(Ain, Bin, BiasP, Cout, Mm, Nnn, Kk, Act)                        \
    gemm_bf16_wmma<<<dim3(((Nnn)+BN-1)/BN, ((Mm)+BM-1)/BM), blk, 0, stream>>>( \
        (Ain), (Bin), (BiasP), (Cout), (Mm), (Nnn), (Kk), (Act))

  // ---------------- GAT layer 1 (4 heads, mean) ----------------
  GEMM(x, W1, (const float*)nullptr, H, N, 4 * HIDC, INDIM, 0);
  hipMemsetAsync(F, 0, (size_t)N * HIDC * sizeof(float), stream);
  gat_alpha<4><<<(N*4 + 255)/256, blk, 0, stream>>>(H, asrc1, adst1, As_, Ad_, Mx, Dn, N);
  gat_edge_max<4><<<edgeBl, blk, 0, stream>>>(ei, E, N, As_, Ad_, Mx);
  gat_fix_max<<<(N*4 + 255)/256, blk, 0, stream>>>(Mx, N * 4);
  gat_edge_sum<4><<<edgeBl, blk, 0, stream>>>(ei, E, N, As_, Ad_, Mx, Dn);
  gat_aggregate<4><<<aggBl, blk, 0, stream>>>(H, ei, E, N, As_, Ad_, Mx, Dn, F);
  bias_bn_relu<<<(N*HIDC + 255)/256, blk, 0, stream>>>(F, b1, bn1g, bn1b, bn1m, bn1v, F, N, HIDC, 1);

  // ---------------- GAT layer 2 (4 heads, mean) ----------------
  GEMM(F, W2, (const float*)nullptr, H, N, 4 * HIDC, HIDC, 0);
  hipMemsetAsync(F, 0, (size_t)N * HIDC * sizeof(float), stream);
  gat_alpha<4><<<(N*4 + 255)/256, blk, 0, stream>>>(H, asrc2, adst2, As_, Ad_, Mx, Dn, N);
  gat_edge_max<4><<<edgeBl, blk, 0, stream>>>(ei, E, N, As_, Ad_, Mx);
  gat_fix_max<<<(N*4 + 255)/256, blk, 0, stream>>>(Mx, N * 4);
  gat_edge_sum<4><<<edgeBl, blk, 0, stream>>>(ei, E, N, As_, Ad_, Mx, Dn);
  gat_aggregate<4><<<aggBl, blk, 0, stream>>>(H, ei, E, N, As_, Ad_, Mx, Dn, F);
  bias_bn_relu<<<(N*HIDC + 255)/256, blk, 0, stream>>>(F, b2, bn2g, bn2b, bn2m, bn2v, F, N, HIDC, 1);

  // ---------------- GAT layer 3 (1 head, concat) ----------------
  GEMM(F, W3, (const float*)nullptr, H, N, HIDC, HIDC, 0);
  hipMemsetAsync(G, 0, (size_t)N * HIDC * sizeof(float), stream);
  gat_alpha<1><<<(N + 255)/256, blk, 0, stream>>>(H, asrc3, adst3, As_, Ad_, Mx, Dn, N);
  gat_edge_max<1><<<edgeBl, blk, 0, stream>>>(ei, E, N, As_, Ad_, Mx);
  gat_fix_max<<<(N + 255)/256, blk, 0, stream>>>(Mx, N);
  gat_edge_sum<1><<<edgeBl, blk, 0, stream>>>(ei, E, N, As_, Ad_, Mx, Dn);
  gat_aggregate<1><<<aggBl, blk, 0, stream>>>(H, ei, E, N, As_, Ad_, Mx, Dn, G);
  bias_bn_relu<<<(N*HIDC + 255)/256, blk, 0, stream>>>(G, b3, nullptr, nullptr, nullptr, nullptr, G, N, HIDC, 0);

  // ---------------- classifier MLP ----------------
  GEMM(G, l1w, l1b, F, N, HIDC, HIDC, 1);            // relu
  GEMM(F, l2w, l2b, H, N, HIDC/2, HIDC, 1);          // relu (H reused as N*64 buf)
  GEMM(H, l3w, l3b, out, N, NCLS, HIDC/2, 2);        // sigmoid -> d_out
  #undef GEMM
}
